// AttentionBlock_89069031785126
// MI455X (gfx1250) — compile-verified, matrix operand order checked by reading, not run on hardware
//
#include <hip/hip_runtime.h>
#include <hip/hip_bf16.h>
#include <stdint.h>

// ---- problem constants (B=2, C=512, H=W=D=32, N=32768, GROUPS=32) ----
#define BATCH 2
#define C_CH  512
#define NTOK  32768
#define NGRP  32
#define CPG   16            // channels per group
#define EPS   1e-5f

typedef __attribute__((ext_vector_type(16))) __bf16 v16bf;
typedef __attribute__((ext_vector_type(8)))  float  v8f;

// ---------------- helpers ----------------
__device__ __forceinline__ unsigned short f2bf(float f) {
  union { float f; unsigned int u; } x; x.f = f;
  unsigned int r = x.u + 0x7FFFu + ((x.u >> 16) & 1u);   // round-to-nearest-even
  return (unsigned short)(r >> 16);
}

// WMMA bf16 A/B fragment load (wave32 layout, ISA 7.12.2):
// lane&15 = row within 16-row tile, lane>>4 selects K-half.
// elements 0..7  = K = k0 + half*8 + 0..7
// elements 8..15 = K = k0 + 16 + half*8 + 0..7
// Matrix stored row-major with K contiguous (ld = K-stride of a row).
__device__ __forceinline__ v16bf load_frag(const unsigned short* __restrict__ base,
                                           int row0, int ld, int k0, int lane) {
  int r    = row0 + (lane & 15);
  int half = lane >> 4;
  const unsigned short* p = base + (size_t)r * (size_t)ld + k0 + half * 8;
  union { uint4 u[2]; v16bf v; } f;
  f.u[0] = *(const uint4*)(p);
  f.u[1] = *(const uint4*)(p + 16);
  return f.v;
}

__device__ __forceinline__ v8f wmma_bf16(v16bf a, v16bf b, v8f c) {
  // (neg_a, A, neg_b, B, c_mod, C, reuse_a, reuse_b)
  return __builtin_amdgcn_wmma_f32_16x16x32_bf16(false, a, false, b, (short)0, c,
                                                 false, false);
}

// 32x32 output tile per wave: 2x2 f32 accumulators, K stepped by 32.
struct Tile32 { v8f c00, c01, c10, c11; };

__device__ __forceinline__ void gemm32x32(const unsigned short* __restrict__ A, int lda,
                                          const unsigned short* __restrict__ B, int ldb,
                                          int m0, int n0, int K, int lane, Tile32& t) {
  t.c00 = {}; t.c01 = {}; t.c10 = {}; t.c11 = {};
  for (int kk = 0; kk < K; kk += 32) {
    v16bf a0 = load_frag(A, m0,      lda, kk, lane);
    v16bf a1 = load_frag(A, m0 + 16, lda, kk, lane);
    v16bf b0 = load_frag(B, n0,      ldb, kk, lane);
    v16bf b1 = load_frag(B, n0 + 16, ldb, kk, lane);
    t.c00 = wmma_bf16(a0, b0, t.c00);
    t.c01 = wmma_bf16(a0, b1, t.c01);
    t.c10 = wmma_bf16(a1, b0, t.c10);
    t.c11 = wmma_bf16(a1, b1, t.c11);
  }
}

// D-matrix element mapping (ISA 7.12.2): VGPR r -> M = r + (lane>=16 ? 8 : 0),
// N = lane & 15.

// ---------------- GroupNorm: per-(b,g) mean / rstd ----------------
// One block per (b,g). Group data is a contiguous slab of CPG*NTOK floats.
__global__ __launch_bounds__(256) void gn_stats_kernel(const float* __restrict__ x,
                                                       float* __restrict__ stats) {
  const size_t GSZ4 = (size_t)CPG * NTOK / 4;   // float4 count
  const float4* p = (const float4*)(x + (size_t)blockIdx.x * CPG * NTOK);
  float s = 0.f, s2 = 0.f;
  for (size_t i = threadIdx.x; i < GSZ4; i += blockDim.x) {
    float4 v = p[i];
    s  += v.x + v.y + v.z + v.w;
    s2 += v.x * v.x + v.y * v.y + v.z * v.z + v.w * v.w;
  }
  __shared__ float sha[256], shb[256];
  sha[threadIdx.x] = s; shb[threadIdx.x] = s2;
  __syncthreads();
  for (int o = 128; o > 0; o >>= 1) {
    if (threadIdx.x < o) {
      sha[threadIdx.x] += sha[threadIdx.x + o];
      shb[threadIdx.x] += shb[threadIdx.x + o];
    }
    __syncthreads();
  }
  if (threadIdx.x == 0) {
    const float inv = 1.0f / (float)((size_t)CPG * NTOK);
    float m   = sha[0] * inv;
    float var = shb[0] * inv - m * m;
    stats[blockIdx.x * 2 + 0] = m;
    stats[blockIdx.x * 2 + 1] = rsqrtf(var + EPS);
  }
}

// ---------------- GroupNorm apply -> xnT [B, N, C] bf16 ----------------
__global__ __launch_bounds__(256) void gn_apply_kernel(const float* __restrict__ x,
                                                       const float* __restrict__ stats,
                                                       const float* __restrict__ gw,
                                                       const float* __restrict__ gb,
                                                       unsigned short* __restrict__ xnT) {
  size_t idx = (size_t)blockIdx.x * blockDim.x + threadIdx.x;  // over B*N*C
  int c = (int)(idx & (C_CH - 1));
  int n = (int)((idx >> 9) & (NTOK - 1));
  int b = (int)(idx >> 24);
  int bg = b * NGRP + (c >> 4);
  float m = stats[bg * 2 + 0];
  float r = stats[bg * 2 + 1];
  float v = x[((size_t)b * C_CH + c) * NTOK + n];
  xnT[idx] = f2bf((v - m) * r * gw[c] + gb[c]);
}

// ---------------- weight f32 -> bf16 ----------------
__global__ __launch_bounds__(256) void wconv_kernel(const float* __restrict__ qkv_w,
                                                    const float* __restrict__ proj_w,
                                                    unsigned short* __restrict__ qkvWb,
                                                    unsigned short* __restrict__ projWb) {
  const size_t NQ = (size_t)3 * C_CH * C_CH;
  size_t idx = (size_t)blockIdx.x * blockDim.x + threadIdx.x;
  if (idx < NQ) qkvWb[idx] = f2bf(qkv_w[idx]);
  else          projWb[idx - NQ] = f2bf(proj_w[idx - NQ]);
}

// ---------------- qkv GEMM: [3C,C] x [C,N] -> q,k [B,C,N], vT [B,N,C] ----------------
__global__ __launch_bounds__(256) void gemm_qkv_kernel(const unsigned short* __restrict__ W,
                                                       const unsigned short* __restrict__ Xt,
                                                       const float* __restrict__ bias,
                                                       unsigned short* __restrict__ q,
                                                       unsigned short* __restrict__ k,
                                                       unsigned short* __restrict__ vT) {
  int b    = blockIdx.z;
  int lane = threadIdx.x & 31;
  int w    = threadIdx.x >> 5;
  int m0   = blockIdx.x * 64  + (w & 1) * 32;
  int n0   = blockIdx.y * 128 + (w >> 1) * 32;
  const unsigned short* B = Xt + (size_t)b * NTOK * C_CH;   // [N, C]
  Tile32 t;
  gemm32x32(W, C_CH, B, C_CH, m0, n0, C_CH, lane, t);

  int col = lane & 15, roff = (lane >> 4) * 8;
  const v8f* accs[4] = { &t.c00, &t.c01, &t.c10, &t.c11 };
  for (int mi = 0; mi < 2; mi++)
    for (int ni = 0; ni < 2; ni++) {
      const v8f a = *accs[mi * 2 + ni];
#pragma unroll
      for (int r = 0; r < 8; r++) {
        int o = m0 + mi * 16 + roff + r;
        int n = n0 + ni * 16 + col;
        unsigned short bv = f2bf(a[r] + bias[o]);
        if (o < C_CH)            q[((size_t)b * C_CH + o) * NTOK + n] = bv;
        else if (o < 2 * C_CH)   k[((size_t)b * C_CH + (o - C_CH)) * NTOK + n] = bv;
        else                     vT[((size_t)b * NTOK + n) * C_CH + (o - 2 * C_CH)] = bv;
      }
    }
}

// ---------------- q.kT GEMM: K = N = 32768 -> attn f32 [B,C,C] ----------------
__global__ __launch_bounds__(256) void gemm_qk_kernel(const unsigned short* __restrict__ q,
                                                      const unsigned short* __restrict__ k,
                                                      float* __restrict__ attn) {
  int b    = blockIdx.z;
  int lane = threadIdx.x & 31;
  int w    = threadIdx.x >> 5;
  int m0   = blockIdx.x * 64  + (w & 1) * 32;
  int n0   = blockIdx.y * 128 + (w >> 1) * 32;
  const unsigned short* A = q + (size_t)b * C_CH * NTOK;
  const unsigned short* B = k + (size_t)b * C_CH * NTOK;
  Tile32 t;
  gemm32x32(A, NTOK, B, NTOK, m0, n0, NTOK, lane, t);

  float* out = attn + (size_t)b * C_CH * C_CH;
  int col = lane & 15, roff = (lane >> 4) * 8;
  const v8f* accs[4] = { &t.c00, &t.c01, &t.c10, &t.c11 };
  for (int mi = 0; mi < 2; mi++)
    for (int ni = 0; ni < 2; ni++) {
      const v8f a = *accs[mi * 2 + ni];
#pragma unroll
      for (int r = 0; r < 8; r++) {
        int cq = m0 + mi * 16 + roff + r;
        int cd = n0 + ni * 16 + col;
        out[(size_t)cq * C_CH + cd] = a[r];
      }
    }
}

// ---------------- softmax over last dim (512), scaled 1/sqrt(C) -> bf16 ----------------
__global__ __launch_bounds__(256) void softmax_kernel(const float* __restrict__ attn,
                                                      unsigned short* __restrict__ attnB) {
  const float scale = 0.04419417382415922f;   // 1/sqrt(512)
  size_t row = (size_t)blockIdx.y * C_CH + blockIdx.x;
  const float* p = attn + row * C_CH;
  unsigned short* o = attnB + row * C_CH;
  int tid = threadIdx.x;
  float v0 = p[tid] * scale, v1 = p[tid + 256] * scale;
  __shared__ float sh[256];
  sh[tid] = fmaxf(v0, v1);
  __syncthreads();
  for (int s = 128; s > 0; s >>= 1) {
    if (tid < s) sh[tid] = fmaxf(sh[tid], sh[tid + s]);
    __syncthreads();
  }
  float m = sh[0];
  __syncthreads();
  float e0 = __expf(v0 - m), e1 = __expf(v1 - m);
  sh[tid] = e0 + e1;
  __syncthreads();
  for (int s = 128; s > 0; s >>= 1) {
    if (tid < s) sh[tid] += sh[tid + s];
    __syncthreads();
  }
  float inv = 1.0f / sh[0];
  o[tid]       = f2bf(e0 * inv);
  o[tid + 256] = f2bf(e1 * inv);
}

// ---------------- attn.v GEMM: [C,C] x [C,N] -> outT [B,N,C] bf16 ----------------
__global__ __launch_bounds__(256) void gemm_av_kernel(const unsigned short* __restrict__ attnB,
                                                      const unsigned short* __restrict__ vT,
                                                      unsigned short* __restrict__ outT) {
  int b    = blockIdx.z;
  int lane = threadIdx.x & 31;
  int w    = threadIdx.x >> 5;
  int m0   = blockIdx.x * 64  + (w & 1) * 32;
  int n0   = blockIdx.y * 128 + (w >> 1) * 32;
  const unsigned short* A = attnB + (size_t)b * C_CH * C_CH;   // [C, C]
  const unsigned short* B = vT    + (size_t)b * NTOK * C_CH;   // [N, C]
  Tile32 t;
  gemm32x32(A, C_CH, B, C_CH, m0, n0, C_CH, lane, t);

  unsigned short* out = outT + (size_t)b * NTOK * C_CH;
  int col = lane & 15, roff = (lane >> 4) * 8;
  const v8f* accs[4] = { &t.c00, &t.c01, &t.c10, &t.c11 };
  for (int mi = 0; mi < 2; mi++)
    for (int ni = 0; ni < 2; ni++) {
      const v8f a = *accs[mi * 2 + ni];
#pragma unroll
      for (int r = 0; r < 8; r++) {
        int c = m0 + mi * 16 + roff + r;
        int n = n0 + ni * 16 + col;
        out[(size_t)n * C_CH + c] = f2bf(a[r]);
      }
    }
}

// ---------------- proj GEMM + bias + residual -> f32 d_out [B,C,N] ----------------
__global__ __launch_bounds__(256) void gemm_proj_kernel(const unsigned short* __restrict__ Wp,
                                                        const unsigned short* __restrict__ outT,
                                                        const float* __restrict__ bias,
                                                        const float* __restrict__ x,
                                                        float* __restrict__ y) {
  int b    = blockIdx.z;
  int lane = threadIdx.x & 31;
  int w    = threadIdx.x >> 5;
  int m0   = blockIdx.x * 64  + (w & 1) * 32;
  int n0   = blockIdx.y * 128 + (w >> 1) * 32;
  const unsigned short* B = outT + (size_t)b * NTOK * C_CH;    // [N, C]
  Tile32 t;
  gemm32x32(Wp, C_CH, B, C_CH, m0, n0, C_CH, lane, t);

  int col = lane & 15, roff = (lane >> 4) * 8;
  const v8f* accs[4] = { &t.c00, &t.c01, &t.c10, &t.c11 };
  for (int mi = 0; mi < 2; mi++)
    for (int ni = 0; ni < 2; ni++) {
      const v8f a = *accs[mi * 2 + ni];
#pragma unroll
      for (int r = 0; r < 8; r++) {
        int o = m0 + mi * 16 + roff + r;
        int n = n0 + ni * 16 + col;
        size_t idx = ((size_t)b * C_CH + o) * NTOK + n;
        y[idx] = x[idx] + a[r] + bias[o];
      }
    }
}

// ---------------- launcher ----------------
extern "C" void kernel_launch(void* const* d_in, const int* in_sizes, int n_in,
                              void* d_out, int out_size, void* d_ws, size_t ws_size,
                              hipStream_t stream) {
  (void)in_sizes; (void)n_in; (void)out_size; (void)ws_size;
  const float* x      = (const float*)d_in[0];
  const float* gn_w   = (const float*)d_in[1];
  const float* gn_b   = (const float*)d_in[2];
  const float* qkv_w  = (const float*)d_in[3];
  const float* qkv_b  = (const float*)d_in[4];
  const float* proj_w = (const float*)d_in[5];
  const float* proj_b = (const float*)d_in[6];
  float* out = (float*)d_out;

  const size_t BNC = (size_t)BATCH * NTOK * C_CH;        // 33.5M elems
  const size_t BCC = (size_t)BATCH * C_CH * C_CH;        // 524K elems

  unsigned short* xnT    = (unsigned short*)d_ws;        // [B,N,C] bf16 (reused as outT)
  unsigned short* qB     = xnT + BNC;                    // [B,C,N]
  unsigned short* kB     = qB + BNC;                     // [B,C,N]
  unsigned short* vT     = kB + BNC;                     // [B,N,C]
  float*          attnF  = (float*)(vT + BNC);           // [B,C,C]
  unsigned short* attnB  = (unsigned short*)(attnF + BCC);
  unsigned short* qkvWb  = attnB + BCC;                  // [3C,C]
  unsigned short* projWb = qkvWb + (size_t)3 * C_CH * C_CH;
  float*          stats  = (float*)(projWb + (size_t)C_CH * C_CH); // [B*G, 2]

  gn_stats_kernel<<<dim3(BATCH * NGRP), dim3(256), 0, stream>>>(x, stats);
  gn_apply_kernel<<<dim3((unsigned)(BNC / 256)), dim3(256), 0, stream>>>(
      x, stats, gn_w, gn_b, xnT);
  wconv_kernel<<<dim3(4096), dim3(256), 0, stream>>>(qkv_w, proj_w, qkvWb, projWb);

  // block tile 64(M) x 128(N), 8 waves of 32x32
  gemm_qkv_kernel<<<dim3(3 * C_CH / 64, NTOK / 128, BATCH), dim3(256), 0, stream>>>(
      qkvWb, xnT, qkv_b, qB, kB, vT);
  gemm_qk_kernel<<<dim3(C_CH / 64, C_CH / 128, BATCH), dim3(256), 0, stream>>>(
      qB, kB, attnF);
  softmax_kernel<<<dim3(C_CH, BATCH), dim3(256), 0, stream>>>(attnF, attnB);
  gemm_av_kernel<<<dim3(C_CH / 64, NTOK / 128, BATCH), dim3(256), 0, stream>>>(
      attnB, vT, xnT /* outT aliases xnT */);
  gemm_proj_kernel<<<dim3(C_CH / 64, NTOK / 128, BATCH), dim3(256), 0, stream>>>(
      projWb, xnT, proj_b, x, out);
}
